// VectorQuantizer_4071628997227
// MI455X (gfx1250) — matrix operand order repaired and checked by reading, not compiled, not run on hardware
//
#include <hip/hip_runtime.h>
#include <hip/hip_bf16.h>

// ---------------------------------------------------------------------------
// VQ-VAE vector quantizer for MI455X (gfx1250, wave32, WMMA + TDM).
//   x:        [32, 64, 64, 64] f32 NCHW  (B=32, C=D=64, H=W=64)
//   codebook: [1024, 64] f32
// Outputs (concat in d_out, float): loss[1], quantized_nchw[8388608], idx[131072]
//
// Distance pass: d_j = |e_j|^2 - 2 x.e_j (|x|^2 row-constant, dropped from
// argmin). GEMM term via split-f16 WMMA: x=xh+xl, e=eh+el (f16 + residual),
// x.e ~= xh*eh + xl*eh + xh*el -> 3x v_wmma_f32_16x16x32_f16 per K=32 chunk,
// argmin-faithful to the f32 reference (~1e-9 dot error vs ~1e-5 code gaps).
// Codebook chunks are DMA'd into LDS by the Tensor Data Mover one chunk ahead
// (double-buffered, TENSORcnt), converted to f16 hi/lo once, then consumed by
// 8 waves x 2 strips of WMMAs. Per-lane argmin is fully branchless: n is
// strictly increasing per lane, so strict '<' gives first-min semantics.
// ---------------------------------------------------------------------------

typedef __attribute__((ext_vector_type(16))) _Float16 v16h;
typedef __attribute__((ext_vector_type(8)))  _Float16 v8h;
typedef __attribute__((ext_vector_type(8)))  float    v8f;
typedef __attribute__((ext_vector_type(4)))  unsigned int v4u;
typedef __attribute__((ext_vector_type(4)))  int      v4i;
typedef __attribute__((ext_vector_type(8)))  int      v8i;

#define VQ_D        64
#define VQ_K        1024
#define VQ_NROWS    131072      // 32*64*64
#define VQ_NUMEL    8388608     // 32*64*64*64
#define VQ_BPAD     88          // f16 LDS row pad (halves): 16B aligned, conflict-free
#define VQ_CHUNK    64          // codes staged per TDM transfer
#define VQ_SPAD     68          // staged f32 row stride (64 + 4 dword TDM pad)

#if defined(__has_builtin)
#if __has_builtin(__builtin_amdgcn_tensor_load_to_lds) && __has_builtin(__builtin_amdgcn_s_wait_tensorcnt)
#define VQ_HAS_TDM 1
#endif
#endif
#ifndef VQ_HAS_TDM
#define VQ_HAS_TDM 0
#endif

// Load 16 contiguous halves (32B) from LDS via two 16B chunks.
static __device__ inline v16h ld16h(const _Float16* p) {
    v8h a = *(const v8h*)(p);
    v8h b = *(const v8h*)(p + 8);
    v16h r;
#pragma unroll
    for (int i = 0; i < 8; ++i) { r[i] = a[i]; r[i + 8] = b[i]; }
    return r;
}

#if VQ_HAS_TDM
// Issue one TDM transfer: 64x64 f32 tile of the codebook -> LDS (row-padded to
// 68 floats via D# pad_interval=64 dwords / pad_amount=4 dwords).
static __device__ inline void tdm_load_chunk(const float* gsrc, unsigned int ldsOff) {
    unsigned long long ga = (unsigned long long)(size_t)gsrc;
    v4u g0;
    g0[0] = 1u;                                   // count=1, user descriptor
    g0[1] = ldsOff;                               // lds_addr (bytes)
    g0[2] = (unsigned int)ga;                     // global_addr[31:0]
    g0[3] = (unsigned int)((ga >> 32) & 0x01FFFFFFu) | 0x80000000u; // ga[56:32]|type=2
    v8i g1;
    g1[0] = (int)((2u << 16) | (1u << 20) | (5u << 22) | (3u << 25));
    //            data_size=4B | pad_enable | pad_interval=64dw | pad_amount=4dw
    g1[1] = (int)(64u << 16);                     // tensor_dim0 = 64  (bits 79:48)
    g1[2] = (int)(1024u << 16);                   // tensor_dim1 = 1024 (bits 111:80)
    g1[3] = (int)(64u << 16);                     // tile_dim0 = 64 (bits 127:112)
    g1[4] = (int)64;                              // tile_dim1 = 64, tile_dim2 = 0
    g1[5] = (int)64;                              // tensor_dim0_stride = 64
    g1[6] = 0;                                    // tensor_dim1_stride = 0 (2D)
    g1[7] = 0;
    v4i z4 = {0, 0, 0, 0};
#if defined(__clang_major__) && (__clang_major__ >= 23)
    v8i z8 = {0, 0, 0, 0, 0, 0, 0, 0};
    __builtin_amdgcn_tensor_load_to_lds(g0, g1, z4, z4, z8, 0);
#else
    __builtin_amdgcn_tensor_load_to_lds(g0, g1, z4, z4, 0);
#endif
}
#endif

// ---------------------------------------------------------------------------
// Kernel 0: esq[k] = |codebook[k]|^2 ; zero the f64 SSE accumulator.
// ---------------------------------------------------------------------------
__global__ __launch_bounds__(256) void vq_init(const float* __restrict__ cb,
                                               float* __restrict__ esq,
                                               double* __restrict__ sseAcc) {
    int k = blockIdx.x * 256 + threadIdx.x;
    if (k == 0) *sseAcc = 0.0;
    if (k < VQ_K) {
        const float4* row = (const float4*)(cb + (size_t)k * VQ_D);
        float s = 0.0f;
#pragma unroll
        for (int i = 0; i < 16; ++i) {
            float4 v = row[i];
            s = fmaf(v.x, v.x, s); s = fmaf(v.y, v.y, s);
            s = fmaf(v.z, v.z, s); s = fmaf(v.w, v.w, s);
        }
        esq[k] = s;
    }
}

// ---------------------------------------------------------------------------
// Kernel 1: distance + argmin. Block = 256 thr (8 waves) -> 256 rows
// (2 strips of 16 per wave). 16 chunks x 4 tiles; 12 WMMAs per tile.
// ---------------------------------------------------------------------------
__global__ __launch_bounds__(256) void vq_argmin(const float* __restrict__ x,
                                                 const float* __restrict__ cb,
                                                 const float* __restrict__ esqG,
                                                 int* __restrict__ idxOut) {
#if VQ_HAS_TDM
    __shared__ float s_stage[2][VQ_CHUNK][VQ_SPAD];   // raw f32, TDM double buffer
#endif
    __shared__ _Float16 s_bh[VQ_CHUNK][VQ_BPAD];      // chunk, f16 hi
    __shared__ _Float16 s_bl[VQ_CHUNK][VQ_BPAD];      // chunk, f16 residual
    __shared__ float    s_esq[VQ_K];                  // |e|^2, resident all pass

    const int tid  = threadIdx.x;
    const int lane = tid & 31;
    const int wave = tid >> 5;
    const int row0 = blockIdx.x * 256 + wave * 32;    // wave's 2x16-row strips

    // stage |e|^2 into LDS (4KB, removes per-tile global load + loadcnt stall)
    ((float4*)s_esq)[tid] = ((const float4*)esqG)[tid];

#if VQ_HAS_TDM
    if (wave == 0) tdm_load_chunk(cb, (unsigned int)(size_t)&s_stage[0][0][0]);
#endif

    // ---- A fragments (contiguous k = khalf + e, same as B), split hi/lo, 2 strips ---
    const int m     = lane & 15;
    const int khalf = (lane >> 4) << 4;               // 0 or 16
    v16h ah[2][2], al[2][2];                          // [strip][K-chunk]
#pragma unroll
    for (int s = 0; s < 2; ++s) {
        const int g  = row0 + s * 16 + m;             // global flat row (NHWC)
        const int bi = g >> 12;
        const int hw = g & 4095;
        const float* xb = x + ((size_t)bi << 18) + hw;
#pragma unroll
        for (int f = 0; f < 2; ++f) {
#pragma unroll
            for (int e = 0; e < 16; ++e) {
                int c = 32 * f + khalf + e;
                float v = xb[(size_t)c << 12];        // coalesced: hw tracks lane
                _Float16 h = (_Float16)v;
                ah[s][f][e] = h;
                al[s][f][e] = (_Float16)(v - (float)h);
            }
        }
    }

    float bestV[2][8];
    int   bestI[2][8];
#pragma unroll
    for (int s = 0; s < 2; ++s)
#pragma unroll
        for (int r = 0; r < 8; ++r) { bestV[s][r] = 3.402823466e38f; bestI[s][r] = 0x7fffffff; }

    const int nl    = lane & 15;                      // column within tile
    const int koffB = (lane >> 4) << 4;               // 0 or 16

    for (int ch = 0; ch < 16; ++ch) {
        const int buf = ch & 1;
#if VQ_HAS_TDM
        __builtin_amdgcn_s_wait_tensorcnt(0);         // chunk ch staged (wave 0)
        __syncthreads();                              // publish + prev tiles done
        if (wave == 0 && ch < 15)                     // prefetch next chunk
            tdm_load_chunk(cb + (size_t)(ch + 1) * (VQ_CHUNK * VQ_D),
                           (unsigned int)(size_t)&s_stage[buf ^ 1][0][0]);
        {   // convert staged f32 -> f16 hi/lo (16 floats per thread)
            int cn = tid >> 2;
            int k0 = (tid & 3) << 4;
#pragma unroll
            for (int j = 0; j < 16; ++j) {
                float v = s_stage[buf][cn][k0 + j];
                _Float16 h = (_Float16)v;
                s_bh[cn][k0 + j] = h;
                s_bl[cn][k0 + j] = (_Float16)(v - (float)h);
            }
        }
#else
        __syncthreads();                              // prev tiles done
        {   // cooperative global load + convert (fallback path)
            int cn = tid >> 2;
            int k0 = (tid & 3) << 4;
            const float4* src = (const float4*)(cb + (size_t)(ch * VQ_CHUNK + cn) * VQ_D + k0);
#pragma unroll
            for (int q4 = 0; q4 < 4; ++q4) {
                float4 v = src[q4];
                float vv[4] = {v.x, v.y, v.z, v.w};
#pragma unroll
                for (int j = 0; j < 4; ++j) {
                    _Float16 h = (_Float16)vv[j];
                    s_bh[cn][k0 + q4 * 4 + j] = h;
                    s_bl[cn][k0 + q4 * 4 + j] = (_Float16)(vv[j] - (float)h);
                }
            }
        }
#endif
        __syncthreads();

#pragma unroll
        for (int tt = 0; tt < 4; ++tt) {              // 4 tiles of 16 codes
            const int cbase = tt * 16;
            v16h bh0 = ld16h(&s_bh[cbase + nl][koffB]);
            v16h bh1 = ld16h(&s_bh[cbase + nl][32 + koffB]);
            v16h bl0 = ld16h(&s_bl[cbase + nl][koffB]);
            v16h bl1 = ld16h(&s_bl[cbase + nl][32 + koffB]);

            v8f acc[2] = {{}, {}};
            acc[0] = __builtin_amdgcn_wmma_f32_16x16x32_f16(false, ah[0][0], false, bh0, (short)0, acc[0], false, false);
            acc[1] = __builtin_amdgcn_wmma_f32_16x16x32_f16(false, ah[1][0], false, bh0, (short)0, acc[1], false, false);
            acc[0] = __builtin_amdgcn_wmma_f32_16x16x32_f16(false, ah[0][1], false, bh1, (short)0, acc[0], false, false);
            acc[1] = __builtin_amdgcn_wmma_f32_16x16x32_f16(false, ah[1][1], false, bh1, (short)0, acc[1], false, false);
            acc[0] = __builtin_amdgcn_wmma_f32_16x16x32_f16(false, al[0][0], false, bh0, (short)0, acc[0], false, false);
            acc[1] = __builtin_amdgcn_wmma_f32_16x16x32_f16(false, al[1][0], false, bh0, (short)0, acc[1], false, false);
            acc[0] = __builtin_amdgcn_wmma_f32_16x16x32_f16(false, al[0][1], false, bh1, (short)0, acc[0], false, false);
            acc[1] = __builtin_amdgcn_wmma_f32_16x16x32_f16(false, al[1][1], false, bh1, (short)0, acc[1], false, false);
            acc[0] = __builtin_amdgcn_wmma_f32_16x16x32_f16(false, ah[0][0], false, bl0, (short)0, acc[0], false, false);
            acc[1] = __builtin_amdgcn_wmma_f32_16x16x32_f16(false, ah[1][0], false, bl0, (short)0, acc[1], false, false);
            acc[0] = __builtin_amdgcn_wmma_f32_16x16x32_f16(false, ah[0][1], false, bl1, (short)0, acc[0], false, false);
            acc[1] = __builtin_amdgcn_wmma_f32_16x16x32_f16(false, ah[1][1], false, bl1, (short)0, acc[1], false, false);

            // Branchless epilogue: d = |e|^2 - 2 x.e. Per lane n strictly
            // increases over the sweep, so strict '<' == argmin first-min
            // tie-break. 1 v_cmp + 2 v_cndmask per element, no exec games.
            const int   n  = ch * VQ_CHUNK + cbase + nl;
            const float e2 = s_esq[n];
#pragma unroll
            for (int s = 0; s < 2; ++s) {
#pragma unroll
                for (int r = 0; r < 8; ++r) {
                    float d = fmaf(-2.0f, acc[s][r], e2);
                    bool better = d < bestV[s][r];
                    bestV[s][r] = better ? d : bestV[s][r];
                    bestI[s][r] = better ? n : bestI[s][r];
                }
            }
        }
    }

    // Cross-lane argmin within each 16-lane group (C layout: lane%16 = column).
    // Here lanes hold different n, so break value-ties toward the smaller idx
    // (reference argmin returns the first minimum). Branchless selects.
#pragma unroll
    for (int mask = 8; mask > 0; mask >>= 1) {
#pragma unroll
        for (int s = 0; s < 2; ++s) {
#pragma unroll
            for (int r = 0; r < 8; ++r) {
                float ov = __shfl_xor(bestV[s][r], mask, 32);
                int   oi = __shfl_xor(bestI[s][r], mask, 32);
                bool better = (ov < bestV[s][r]) |
                              ((ov == bestV[s][r]) & (oi < bestI[s][r]));
                bestV[s][r] = better ? ov : bestV[s][r];
                bestI[s][r] = better ? oi : bestI[s][r];
            }
        }
    }
    if ((lane & 15) == 0) {        // lane 0 -> rows 0..7, lane 16 -> rows 8..15
#pragma unroll
        for (int s = 0; s < 2; ++s) {
            int rbase = row0 + s * 16 + ((lane >> 4) << 3);
#pragma unroll
            for (int r = 0; r < 8; ++r) idxOut[rbase + r] = bestI[s][r];
        }
    }
}

// ---------------------------------------------------------------------------
// Kernel 2: gather codebook[idx], write quantized NCHW (coalesced: hw = tid),
// emit idx as float, accumulate SSE into f64 (global_atomic_add_f64).
// ---------------------------------------------------------------------------
__global__ __launch_bounds__(256) void vq_gather(const float* __restrict__ x,
                                                 const float* __restrict__ cb,
                                                 const int* __restrict__ idxIn,
                                                 float* __restrict__ outQ,
                                                 float* __restrict__ outIdx,
                                                 double* __restrict__ sseAcc) {
    const int tid = threadIdx.x;
    const int row = blockIdx.x * 256 + tid;
    const int bi  = row >> 12;
    const int hw  = row & 4095;
    const int id  = idxIn[row];
    outIdx[row]   = (float)id;

    const float* crow = cb + (size_t)id * VQ_D;      // L2-resident gather
    const float* xb   = x    + ((size_t)bi << 18) + hw;
    float*       ob   = outQ + ((size_t)bi << 18) + hw;

    float sse = 0.0f;
#pragma unroll 8
    for (int c = 0; c < VQ_D; ++c) {
        float q  = crow[c];
        float xv = xb[(size_t)c << 12];
        float d  = q - xv;
        sse = fmaf(d, d, sse);
        ob[(size_t)c << 12] = q;                     // coalesced across threads
    }
#pragma unroll
    for (int off = 16; off > 0; off >>= 1) sse += __shfl_xor(sse, off, 32);
    if ((tid & 31) == 0) atomicAdd(sseAcc, (double)sse);
}

// ---------------------------------------------------------------------------
// Kernel 3: loss = (1 + 0.25) * SSE / numel   (e_latent == q_latent forward)
// ---------------------------------------------------------------------------
__global__ void vq_loss(const double* __restrict__ sseAcc, float* __restrict__ out) {
    out[0] = (float)(1.25 * (*sseAcc) * (1.0 / (double)VQ_NUMEL));
}

// ---------------------------------------------------------------------------
extern "C" void kernel_launch(void* const* d_in, const int* in_sizes, int n_in,
                              void* d_out, int out_size, void* d_ws, size_t ws_size,
                              hipStream_t stream) {
    (void)in_sizes; (void)n_in; (void)out_size; (void)ws_size;
    const float* x  = (const float*)d_in[0];
    const float* cb = (const float*)d_in[1];

    float*  out     = (float*)d_out;
    float*  outLoss = out;                       // [1]
    float*  outQ    = out + 1;                   // [8388608]
    float*  outIdx  = out + 1 + VQ_NUMEL;        // [131072]

    // scratch layout: esq[1024] f32 | f64 SSE | int idx[131072]
    float*  esq    = (float*)d_ws;
    double* sseAcc = (double*)((char*)d_ws + 4096);
    int*    idxInt = (int*)((char*)d_ws + 4352);

    vq_init  <<<4, 256, 0, stream>>>(cb, esq, sseAcc);
    vq_argmin<<<VQ_NROWS / 256, 256, 0, stream>>>(x, cb, esq, idxInt);
    vq_gather<<<VQ_NROWS / 256, 256, 0, stream>>>(x, cb, idxInt, outQ, outIdx, sseAcc);
    vq_loss  <<<1, 1, 0, stream>>>(sseAcc, outLoss);
}
